// Llama_75960791597331
// MI455X (gfx1250) — matrix operand order repaired and checked by reading, not compiled, not run on hardware
//
#include <hip/hip_runtime.h>
#include <math.h>

// ---------------- model constants (match reference) ----------------
#define B_    2
#define T_    1024
#define D_    1024
#define H_    16
#define KVH_  4
#define HD_   64
#define L_    4
#define I_    2730
#define V_    32000
#define KVW_  (2 * D_ / (H_ / KVH_))   // 512 = concat(k,v) row width
#define M_    (B_ * T_)                // 2048 token rows

// ---------------- CDNA5 WMMA types ----------------
typedef __attribute__((ext_vector_type(16))) __bf16 v16bf;
typedef __attribute__((ext_vector_type(8)))  float  v8f;

// A frag (16x32 bf16): lane half h, lane&15 = row m.
//   halfwords 0..7  -> K = k0 + 8h + i ; halfwords 8..15 -> K = k0 + 16 + 8h + i
// All rows are 8B aligned (K even in every call) -> float2 loads, no guards.
__device__ __forceinline__ v16bf load_a_frag(const float* __restrict__ row,
                                             int k0, int half) {
  const float2* p0 = (const float2*)(row + k0 + half * 8);
  const float2* p1 = (const float2*)(row + k0 + 16 + half * 8);
  v16bf v;
#pragma unroll
  for (int i = 0; i < 4; ++i) {
    const float2 x0 = p0[i], x1 = p1[i];
    v[2 * i]         = (__bf16)x0.x;  v[2 * i + 1]     = (__bf16)x0.y;
    v[8 + 2 * i]     = (__bf16)x1.x;  v[8 + 2 * i + 1] = (__bf16)x1.y;
  }
  return v;
}

// B frag (32x16 bf16): lane&15 = col n; halfwords 0..7 -> K = k0 + 8h + i ; 8..15 -> K = k0 + 16 + 8h + i
__device__ __forceinline__ v16bf load_b_frag(const float* __restrict__ row,
                                             int k0, int half) {
  const float2* p0 = (const float2*)(row + k0 + half * 8);
  const float2* p1 = (const float2*)(row + k0 + 16 + half * 8);
  v16bf v;
#pragma unroll
  for (int i = 0; i < 4; ++i) {
    const float2 x0 = p0[i], x1 = p1[i];
    v[2 * i]     = (__bf16)x0.x;  v[2 * i + 1]     = (__bf16)x0.y;
    v[8 + 2 * i] = (__bf16)x1.x;  v[8 + 2 * i + 1] = (__bf16)x1.y;
  }
  return v;
}

// Guarded (zero-filled) variants, used only for the final K tail (K=2730).
__device__ __forceinline__ v16bf load_a_frag_tail(const float* __restrict__ row,
                                                  int k0, int half, int K) {
  v16bf v;
#pragma unroll
  for (int i = 0; i < 8; ++i) {
    const int k1 = k0 + half * 8 + i;
    const int k2 = k1 + 16;
    v[i]     = (__bf16)(k1 < K ? row[k1] : 0.0f);
    v[i + 8] = (__bf16)(k2 < K ? row[k2] : 0.0f);
  }
  return v;
}
__device__ __forceinline__ v16bf load_b_frag_tail(const float* __restrict__ row,
                                                  int k0, int half, int K) {
  v16bf v;
#pragma unroll
  for (int i = 0; i < 16; ++i) {
    const int kk = k0 + half * 8 + (i & 7) + ((i >> 3) << 4);
    v[i] = (__bf16)(kk < K ? row[kk] : 0.0f);
  }
  return v;
}

// =====================================================================
// GEMM: C[M,N] = A[M,K] (row major) * W[N,K]^T, fp32 in/out, bf16 WMMA.
// One wave -> 32(M) x 64(N) tile: 8 x v_wmma_f32_16x16x32_bf16 per K step.
// OOB A rows / B cols only feed unstored outputs -> clamp pointers, no zeroing;
// only the K tail zero-fills. Blocks are N-major so the 8MB activation matrix
// stays resident in the 192MB L2 while weights stream from HBM exactly once.
// =====================================================================
__global__ void __launch_bounds__(256)
gemm_bf16_wmma_kernel(const float* __restrict__ A, const float* __restrict__ W,
                      float* __restrict__ C, int M, int N, int K) {
  const int lane = threadIdx.x & 31;
  const int wave = threadIdx.x >> 5;
  const int gw   = blockIdx.x * 8 + wave;
  const int mTiles  = (M + 31) >> 5;
  const int nStrips = (N + 63) >> 6;
  if (gw >= mTiles * nStrips) return;
  const int mT = gw % mTiles;          // N-major ordering: A stays in L2
  const int nS = gw / mTiles;

  const int half = lane >> 4;
  const int l15  = lane & 15;

  const int m0 = mT * 32 + l15;
  const int m1 = m0 + 16;
  const float* Arow0 = A + (size_t)(m0 < M ? m0 : 0) * K;
  const float* Arow1 = A + (size_t)(m1 < M ? m1 : 0) * K;
  const float* Wrow[4];
#pragma unroll
  for (int j = 0; j < 4; ++j) {
    const int n = nS * 64 + j * 16 + l15;
    Wrow[j] = W + (size_t)(n < N ? n : 0) * K;
  }

  v8f acc[8] = {};   // acc[j] = tile(m0..m0+15), acc[4+j] = tile(m1..m1+15)

  const int Kmain = K & ~31;
  for (int k0 = 0; k0 < Kmain; k0 += 32) {
    __builtin_prefetch(Arow0 + k0 + 128, 0, 1);   // global_prefetch_b8
    __builtin_prefetch(Arow1 + k0 + 128, 0, 1);
    const v16bf a0 = load_a_frag(Arow0, k0, half);
    const v16bf a1 = load_a_frag(Arow1, k0, half);
#pragma unroll
    for (int j = 0; j < 4; ++j) {
      __builtin_prefetch(Wrow[j] + k0 + 128, 0, 1);
      const v16bf b = load_b_frag(Wrow[j], k0, half);
      acc[j]     = __builtin_amdgcn_wmma_f32_16x16x32_bf16(
          false, a0, false, b, (short)0, acc[j], false, false);
      acc[4 + j] = __builtin_amdgcn_wmma_f32_16x16x32_bf16(
          false, a1, false, b, (short)0, acc[4 + j], false, false);
    }
  }
  if (Kmain < K) {                      // single zero-filled tail step
    const v16bf a0 = load_a_frag_tail(Arow0, Kmain, half, K);
    const v16bf a1 = load_a_frag_tail(Arow1, Kmain, half, K);
#pragma unroll
    for (int j = 0; j < 4; ++j) {
      const v16bf b = load_b_frag_tail(Wrow[j], Kmain, half, K);
      acc[j]     = __builtin_amdgcn_wmma_f32_16x16x32_bf16(
          false, a0, false, b, (short)0, acc[j], false, false);
      acc[4 + j] = __builtin_amdgcn_wmma_f32_16x16x32_bf16(
          false, a1, false, b, (short)0, acc[4 + j], false, false);
    }
  }

  // C/D frag: vgpr r, lane half h, lane&15 = n -> M = tileBase + 8h + r
#pragma unroll
  for (int mi = 0; mi < 2; ++mi) {
#pragma unroll
    for (int j = 0; j < 4; ++j) {
      const int n = nS * 64 + j * 16 + l15;
      if (n < N) {
#pragma unroll
        for (int r = 0; r < 8; ++r) {
          const int mm = mT * 32 + mi * 16 + half * 8 + r;
          if (mm < M) C[(size_t)mm * N + n] = acc[mi * 4 + j][r];
        }
      }
    }
  }
}

// =====================================================================
// Embedding gather: x[b,t,:] = wte[idx[b,t],:]
// =====================================================================
__global__ void __launch_bounds__(256)
embed_kernel(const int* __restrict__ idx, const float* __restrict__ wte,
             float* __restrict__ x, int total) {
  const int g = blockIdx.x * 256 + threadIdx.x;
  if (g >= total) return;
  const int row = g >> 10;             // D_ = 1024
  const int col = g & 1023;
  x[g] = wte[(size_t)idx[row] * D_ + col];
}

// =====================================================================
// RMSNorm: one block per token row (D = 1024)
// =====================================================================
__global__ void __launch_bounds__(256)
rmsnorm_kernel(const float* __restrict__ x, const float* __restrict__ w,
               float* __restrict__ out) {
  const int row = blockIdx.x;
  const float* xr = x + (size_t)row * D_;
  float ss = 0.0f;
  for (int i = threadIdx.x; i < D_; i += 256) { const float v = xr[i]; ss += v * v; }
  for (int off = 16; off; off >>= 1) ss += __shfl_xor(ss, off);
  __shared__ float red[8];
  __shared__ float stot;
  if ((threadIdx.x & 31) == 0) red[threadIdx.x >> 5] = ss;
  __syncthreads();
  if (threadIdx.x == 0) {
    float t = 0.0f;
    for (int i = 0; i < 8; ++i) t += red[i];
    stot = t;
  }
  __syncthreads();
  const float inv = rsqrtf(stot * (1.0f / D_) + 1e-6f);
  for (int i = threadIdx.x; i < D_; i += 256)
    out[(size_t)row * D_ + i] = w[i] * (xr[i] * inv);
}

// =====================================================================
// RoPE in place (GPT-NeoX pairing): for d in [0,32), pair (d, d+32) of each
// head rotates by angle t / BASE^(d/32). One thread per pair.
// =====================================================================
__global__ void __launch_bounds__(256)
rope_kernel(float* __restrict__ buf, int nHeads, int rowStride, int total) {
  const int g = blockIdx.x * 256 + threadIdx.x;
  if (g >= total) return;
  const int d   = g & 31;
  const int rem = g >> 5;
  const int h   = rem % nHeads;
  const int bt  = rem / nHeads;
  const int t   = bt & (T_ - 1);
  float* p = buf + (size_t)bt * rowStride + h * HD_;
  const float inv_freq = expf(-(float)d * (9.2103403719761840f / 32.0f)); // ln(1e4)/32
  const float ang = (float)t * inv_freq;
  const float s = sinf(ang), c = cosf(ang);
  const float lo = p[d], hi = p[d + 32];
  p[d]      = lo * c - hi * s;
  p[d + 32] = hi * c + lo * s;
}

// =====================================================================
// Causal attention, GQA (q head h reads kv head h%4). q/k already RoPE'd.
// One wave per (b,h,qpos); 3-pass softmax with per-wave LDS score row.
// =====================================================================
__global__ void __launch_bounds__(256)
attn_kernel(const float* __restrict__ q, const float* __restrict__ kv,
            float* __restrict__ y) {
  __shared__ float sc[8][T_];
  __shared__ float qs[8][HD_];
  const int wave = threadIdx.x >> 5;
  const int lane = threadIdx.x & 31;
  const int gw = blockIdx.x * 8 + wave;
  const int t   = gw % T_;
  const int rem = gw / T_;
  const int h   = rem % H_;
  const int b   = rem / H_;

  const float* qrow = q + ((size_t)(b * T_ + t) * D_) + h * HD_;
  qs[wave][lane]      = qrow[lane]      * 0.125f;   // 1/sqrt(64) folded into q
  qs[wave][lane + 32] = qrow[lane + 32] * 0.125f;
  __builtin_amdgcn_wave_barrier();

  const int hk = h & (KVH_ - 1);
  const size_t bt0 = (size_t)b * T_;
  const float* kbase = kv + hk * HD_;
  const float* vbase = kv + KVH_ * HD_ + hk * HD_;   // v half starts at col 256

  // pass 1: scores + running max (lane-per-key)
  float mloc = -INFINITY;
  for (int base = 0; base <= t; base += 32) {
    const int key = base + lane;
    if (key <= t) {
      const float4* kr4 = (const float4*)(kbase + (bt0 + key) * KVW_);
      float acc = 0.0f;
#pragma unroll
      for (int dd = 0; dd < 16; ++dd) {
        const float4 kk = kr4[dd];
        acc += qs[wave][4 * dd + 0] * kk.x + qs[wave][4 * dd + 1] * kk.y +
               qs[wave][4 * dd + 2] * kk.z + qs[wave][4 * dd + 3] * kk.w;
      }
      sc[wave][key] = acc;
      mloc = fmaxf(mloc, acc);
    }
  }
  for (int off = 16; off; off >>= 1) mloc = fmaxf(mloc, __shfl_xor(mloc, off));

  // pass 2: exp + sum
  float ssum = 0.0f;
  for (int base = 0; base <= t; base += 32) {
    const int key = base + lane;
    if (key <= t) {
      const float p = expf(sc[wave][key] - mloc);
      sc[wave][key] = p;
      ssum += p;
    }
  }
  for (int off = 16; off; off >>= 1) ssum += __shfl_xor(ssum, off);
  const float invs = 1.0f / ssum;
  __builtin_amdgcn_wave_barrier();

  // pass 3: y = P @ V (lane-per-dim, 2 dims per lane)
  float a0 = 0.0f, a1 = 0.0f;
  for (int t2 = 0; t2 <= t; ++t2) {
    const float p = sc[wave][t2];
    const float* vr = vbase + (bt0 + t2) * KVW_;
    a0 += p * vr[lane];
    a1 += p * vr[lane + 32];
  }
  float* yrow = y + ((size_t)(b * T_ + t) * D_) + h * HD_;
  yrow[lane]      = a0 * invs;
  yrow[lane + 32] = a1 * invs;
}

// =====================================================================
// Elementwise: g = silu(g) * u ; x += r
// =====================================================================
__global__ void __launch_bounds__(256)
silu_mul_kernel(float* __restrict__ g, const float* __restrict__ u, int n) {
  const int i = blockIdx.x * 256 + threadIdx.x;
  if (i >= n) return;
  const float v = g[i];
  g[i] = (v / (1.0f + expf(-v))) * u[i];
}

__global__ void __launch_bounds__(256)
add_kernel(float* __restrict__ x, const float* __restrict__ r, int n) {
  const int i = blockIdx.x * 256 + threadIdx.x;
  if (i >= n) return;
  x[i] += r[i];
}

// =====================================================================
// Host-side orchestration (all launches on `stream`, graph-capture safe)
// =====================================================================
static inline void launch_gemm(const float* A, const float* W, float* C,
                               int M, int N, int K, hipStream_t stream) {
  const int waves = ((M + 31) / 32) * ((N + 63) / 64);
  gemm_bf16_wmma_kernel<<<(waves + 7) / 8, 256, 0, stream>>>(A, W, C, M, N, K);
}

extern "C" void kernel_launch(void* const* d_in, const int* in_sizes, int n_in,
                              void* d_out, int out_size, void* d_ws, size_t ws_size,
                              hipStream_t stream) {
  (void)in_sizes; (void)n_in; (void)out_size; (void)ws_size;
  const int*   idx  = (const int*)  d_in[0];
  const float* wte  = (const float*)d_in[1];
  const float* Wq   = (const float*)d_in[2];
  const float* Wkv  = (const float*)d_in[3];
  const float* Wo   = (const float*)d_in[4];
  const float* Wg   = (const float*)d_in[5];
  const float* Wu   = (const float*)d_in[6];
  const float* Wd   = (const float*)d_in[7];
  const float* rms1 = (const float*)d_in[8];
  const float* rms2 = (const float*)d_in[9];
  const float* rmsf = (const float*)d_in[10];
  float* out = (float*)d_out;

  // workspace carve-out (~91 MB of f32)
  float* x    = (float*)d_ws;                 // [M, D]
  float* xn   = x    + (size_t)M_ * D_;       // [M, D]
  float* q    = xn   + (size_t)M_ * D_;       // [M, D]
  float* kvb  = q    + (size_t)M_ * D_;       // [M, 512] (k | v)
  float* yb   = kvb  + (size_t)M_ * KVW_;     // [M, D]
  float* proj = yb   + (size_t)M_ * D_;       // [M, D]
  float* g    = proj + (size_t)M_ * D_;       // [M, I]
  float* u    = g    + (size_t)M_ * I_;       // [M, I]

  embed_kernel<<<(M_ * D_ + 255) / 256, 256, 0, stream>>>(idx, wte, x, M_ * D_);

  for (int l = 0; l < L_; ++l) {
    rmsnorm_kernel<<<M_, 256, 0, stream>>>(x, rms1 + (size_t)l * D_, xn);
    launch_gemm(xn, Wq  + (size_t)l * D_ * D_,   q,   M_, D_,   D_, stream);
    launch_gemm(xn, Wkv + (size_t)l * KVW_ * D_, kvb, M_, KVW_, D_, stream);
    {
      const int tq = M_ * H_ * 32;
      rope_kernel<<<(tq + 255) / 256, 256, 0, stream>>>(q, H_, D_, tq);
      const int tk = M_ * KVH_ * 32;
      rope_kernel<<<(tk + 255) / 256, 256, 0, stream>>>(kvb, KVH_, KVW_, tk);
    }
    attn_kernel<<<(B_ * H_ * T_) / 8, 256, 0, stream>>>(q, kvb, yb);
    launch_gemm(yb, Wo + (size_t)l * D_ * D_, proj, M_, D_, D_, stream);
    add_kernel<<<(M_ * D_ + 255) / 256, 256, 0, stream>>>(x, proj, M_ * D_);

    rmsnorm_kernel<<<M_, 256, 0, stream>>>(x, rms2 + (size_t)l * D_, xn);
    launch_gemm(xn, Wg + (size_t)l * I_ * D_, g, M_, I_, D_, stream);
    launch_gemm(xn, Wu + (size_t)l * I_ * D_, u, M_, I_, D_, stream);
    silu_mul_kernel<<<(M_ * I_ + 255) / 256, 256, 0, stream>>>(g, u, M_ * I_);
    launch_gemm(g, Wd + (size_t)l * D_ * I_, proj, M_, D_, I_, stream);
    add_kernel<<<(M_ * D_ + 255) / 256, 256, 0, stream>>>(x, proj, M_ * D_);
  }

  rmsnorm_kernel<<<M_, 256, 0, stream>>>(x, rmsf, xn);
  launch_gemm(xn, wte, out, M_, V_, D_, stream);   // tied lm_head -> logits
}
